// CoLALinear_52364241273350
// MI455X (gfx1250) — matrix operand order repaired and checked by reading, not compiled, not run on hardware
//
#include <hip/hip_runtime.h>
#include <hip/hip_bf16.h>

typedef __attribute__((ext_vector_type(16))) _Float16 v16h;
typedef __attribute__((ext_vector_type(8)))  _Float16 v8h;
typedef __attribute__((ext_vector_type(8)))  float    v8f;
typedef __attribute__((ext_vector_type(4)))  float    v4f;

// d_in = 64*64, tokens are rows of 4096. Per token:
//   T  = Xr(64x64) @ V(64x64)          (A=Xr, B=V)
//   Y^T= T^T(64x64) @ U(64x64)         (A=T^T, B=U),  y[d] with d = k*64 + l
//
// WMMA f32_16x16x32_f16 fragment layouts (wave32, from CDNA5 ISA):
//   A (16x32, M x K): lane L -> M = L%16, half = L/16;
//       element e (0..15): K = (e>>3)*16 + half*8 + (e&7)
//   B (32x16, K x N): lane L -> N = L%16, half = L/16;
//       element e: K = (e>>3)*16 + half*8 + (e&7)
//   C/D (16x16 f32): lane L, reg r: row M = half*8 + r, col N = L%16
//
// The C/D layout of T converts lane-locally into the A layout of T^T:
//   A'frag[lt][q].e  = f16( Tacc[2q + (e>>3)][lt].reg[e&7] )   (same lane)

__global__ __launch_bounds__(128) void cola_kron_kernel(
    const float* __restrict__ x, const float* __restrict__ U,
    const float* __restrict__ V, const float* __restrict__ bvec,
    float* __restrict__ y, int ntok)
{
    __shared__ __align__(16) _Float16 sVB[8 * 32 * 16];   // V B-fragments (8 KB)
    __shared__ __align__(16) _Float16 sUB[8 * 32 * 16];   // U B-fragments (8 KB)
    __shared__ __align__(16) float    sB[4096];           // bias (16 KB)
    __shared__ __align__(16) _Float16 sT[4][8 * 32 * 16]; // per-wave T^T A-fragments (32 KB)

    const int lane = threadIdx.x & 31;
    const int wave = threadIdx.x >> 5;
    const int half = lane >> 4;       // 0 or 1
    const int lm   = lane & 15;       // 0..15

    // ---- one-time per-block setup: build B-fragments for V and U, stage bias ----
    if (wave == 0) {
        #pragma unroll
        for (int f = 0; f < 8; ++f) {                 // f = lt*2 + q
            const int lt = f >> 1, q = f & 1;
            const int l = lt * 16 + lm;
            const int jb = q * 32 + half * 8;
            #pragma unroll
            for (int e = 0; e < 16; ++e)
                sVB[(f * 32 + lane) * 16 + e] = (_Float16)V[(jb + (e & 7) + ((e >> 3) << 4)) * 64 + l];
        }
    } else if (wave == 1) {
        #pragma unroll
        for (int f = 0; f < 8; ++f) {                 // f = kt*2 + q
            const int kt = f >> 1, q = f & 1;
            const int k = kt * 16 + lm;
            const int ib = q * 32 + half * 8;
            #pragma unroll
            for (int e = 0; e < 16; ++e)
                sUB[(f * 32 + lane) * 16 + e] = (_Float16)U[(ib + (e & 7) + ((e >> 3) << 4)) * 64 + k];
        }
    }
    for (int idx = threadIdx.x; idx < 4096; idx += 128) sB[idx] = bvec[idx];
    __syncthreads();

    const int gwave  = blockIdx.x * 4 + wave;
    const int nwaves = gridDim.x * 4;

    for (int tok = gwave; tok < ntok; tok += nwaves) {
        const float* xt = x + (size_t)tok * 4096;
        float*       yt = y + (size_t)tok * 4096;

        // ---------------- stage 1: T = Xr @ V ----------------
        #pragma unroll
        for (int it = 0; it < 4; ++it) {
            v8f acc0 = {}, acc1 = {}, acc2 = {}, acc3 = {};
            #pragma unroll
            for (int q = 0; q < 2; ++q) {
                // A fragment of Xr, tile row it, K-chunk q: two 8-float runs
                const float* xr = xt + (it * 16 + lm) * 64 + q * 32 + half * 8;
                v4f a0 = *(const v4f*)(xr +  0);
                v4f a1 = *(const v4f*)(xr +  4);
                v4f a2 = *(const v4f*)(xr + 16);
                v4f a3 = *(const v4f*)(xr + 20);
                v16h A;
                #pragma unroll
                for (int u = 0; u < 4; ++u) {
                    A[u]      = (_Float16)a0[u];
                    A[4 + u]  = (_Float16)a1[u];
                    A[8 + u]  = (_Float16)a2[u];
                    A[12 + u] = (_Float16)a3[u];
                }
                const v16h B0 = *(const v16h*)&sVB[((0 * 2 + q) * 32 + lane) * 16];
                const v16h B1 = *(const v16h*)&sVB[((1 * 2 + q) * 32 + lane) * 16];
                const v16h B2 = *(const v16h*)&sVB[((2 * 2 + q) * 32 + lane) * 16];
                const v16h B3 = *(const v16h*)&sVB[((3 * 2 + q) * 32 + lane) * 16];
                acc0 = __builtin_amdgcn_wmma_f32_16x16x32_f16(false, A, false, B0, (short)0, acc0, false, false);
                acc1 = __builtin_amdgcn_wmma_f32_16x16x32_f16(false, A, false, B1, (short)0, acc1, false, false);
                acc2 = __builtin_amdgcn_wmma_f32_16x16x32_f16(false, A, false, B2, (short)0, acc2, false, false);
                acc3 = __builtin_amdgcn_wmma_f32_16x16x32_f16(false, A, false, B3, (short)0, acc3, false, false);
            }
            // Convert each 16x16 T tile lane-locally into half of a T^T A-fragment
            v8f accs[4] = {acc0, acc1, acc2, acc3};
            #pragma unroll
            for (int lt = 0; lt < 4; ++lt) {
                v8h h;
                #pragma unroll
                for (int r = 0; r < 8; ++r) h[r] = (_Float16)accs[lt][r];
                const int frag = lt * 2 + (it >> 1);
                *(v8h*)&sT[wave][(frag * 32 + lane) * 16 + (it & 1) * 8] = h;
            }
        }
        // per-wave buffer, LDS ops from one wave are in order: no barrier needed

        // ---------------- stage 2: Y^T = T^T @ U (+ bias), store ----------------
        #pragma unroll
        for (int lt = 0; lt < 4; ++lt) {
            const v16h A0 = *(const v16h*)&sT[wave][((lt * 2 + 0) * 32 + lane) * 16];
            const v16h A1 = *(const v16h*)&sT[wave][((lt * 2 + 1) * 32 + lane) * 16];
            #pragma unroll
            for (int kt = 0; kt < 4; ++kt) {
                v8f acc = {};
                const v16h B0 = *(const v16h*)&sUB[((kt * 2 + 0) * 32 + lane) * 16];
                const v16h B1 = *(const v16h*)&sUB[((kt * 2 + 1) * 32 + lane) * 16];
                acc = __builtin_amdgcn_wmma_f32_16x16x32_f16(false, A0, false, B0, (short)0, acc, false, false);
                acc = __builtin_amdgcn_wmma_f32_16x16x32_f16(false, A1, false, B1, (short)0, acc, false, false);
                // lane holds y rows d = (kt*16+lm)*64 + lt*16 + half*8 + r, r=0..7 contiguous
                const int dbase = (kt * 16 + lm) * 64 + lt * 16 + half * 8;
                v4f o0, o1;
                #pragma unroll
                for (int r = 0; r < 4; ++r) {
                    o0[r] = acc[r]     + sB[dbase + r];
                    o1[r] = acc[4 + r] + sB[dbase + 4 + r];
                }
                *(v4f*)(yt + dbase)     = o0;
                *(v4f*)(yt + dbase + 4) = o1;
            }
        }
    }
}

extern "C" void kernel_launch(void* const* d_in, const int* in_sizes, int n_in,
                              void* d_out, int out_size, void* d_ws, size_t ws_size,
                              hipStream_t stream) {
    const float* x = (const float*)d_in[0];
    const float* U = (const float*)d_in[1];
    const float* V = (const float*)d_in[2];
    const float* b = (const float*)d_in[3];
    float* y = (float*)d_out;
    const int ntok = in_sizes[0] / 4096;   // 16384 for [4,4096,4096]

    int blocks = (ntok + 15) / 16;         // 4 waves/block, ~4 tokens per wave
    if (blocks > 1024) blocks = 1024;
    if (blocks < 1) blocks = 1;
    hipLaunchKernelGGL(cola_kron_kernel, dim3(blocks), dim3(128), 0, stream,
                       x, U, V, b, y, ntok);
}
